// AttentionLayer_49039936586058
// MI455X (gfx1250) — matrix operand order repaired and checked by reading, not compiled, not run on hardware
//
#include <hip/hip_runtime.h>
#include <stdint.h>

// ---------------- problem constants ----------------
#define B_   4096
#define S_   200
#define D_   64
#define H1_  64
#define H2_  32
#define K1_  256        // 4*D
#define NT_  13         // ceil(S/16) s-tiles

// ---------------- LDS strides (padded for bank spread) ----------------
#define KPAD  68        // keys row stride (f32 elems)
#define APAD  264       // att_in / W1 row stride (bf16 elems)
#define W2PAD 72        // W2 row stride (bf16 elems)
#define HPAD  72        // h row stride (bf16 elems)
#define SCPAD 33        // score tile row stride (f32 elems)

typedef __attribute__((ext_vector_type(16))) __bf16 v16bf;
typedef __attribute__((ext_vector_type(8)))  float  v8f;

union AFrag { v16bf v; uint32_t u[8]; };

__device__ __forceinline__ uint16_t f2bf(float x) {
  uint32_t u = __builtin_bit_cast(uint32_t, x);
  u += 0x7fffu + ((u >> 16) & 1u);          // round-to-nearest-even
  return (uint16_t)(u >> 16);
}

__device__ __forceinline__ uint32_t ld2bf(const uint16_t* p) {
  return *(const uint32_t*)p;               // p is 4B aligned by construction
}

__global__ __launch_bounds__(128)
void din_attn_kernel(const float* __restrict__ query,
                     const float* __restrict__ keys,
                     const int*   __restrict__ kmask,
                     const float* __restrict__ W1, const float* __restrict__ b1,
                     const float* __restrict__ a1,
                     const float* __restrict__ W2, const float* __restrict__ b2,
                     const float* __restrict__ a2,
                     const float* __restrict__ W3, const float* __restrict__ b3,
                     float* __restrict__ outSum,   // [B, D]
                     float* __restrict__ outW)     // [B, S]
{
  extern __shared__ char smem[];
  float*    keysLds = (float*)smem;              // [200*KPAD] f32 keys (kept for weighted sum)
  float*    scTile  = keysLds + S_ * KPAD;       // [16*SCPAD] h2 tile
  float*    scores  = scTile + 16 * SCPAD;       // [208] masked scores -> exp -> weights
  float*    qLds    = scores + 208;              // [64]
  float*    b1Lds   = qLds + 64;                 // [64]
  float*    b2Lds   = b1Lds + 64;                // [32]
  float*    w3Lds   = b2Lds + 32;                // [32]
  float*    red     = w3Lds + 32;                // [128] reductions
  uint16_t* w1Lds   = (uint16_t*)(red + 128);    // [64][APAD] bf16, W1^T (col-major by n)
  uint16_t* w2Lds   = w1Lds + H1_ * APAD;        // [32][W2PAD] bf16, W2^T
  uint16_t* attLds  = w2Lds + H2_ * W2PAD;       // [16][APAD] bf16 att_in tile
  uint16_t* hLds    = attLds + 16 * APAD;        // [16][HPAD] bf16 h tile

  const int tid    = threadIdx.x;
  const int wave   = tid >> 5;
  const int lane   = tid & 31;
  const int ln16   = lane & 15;
  const int laneHi = (lane >= 16) ? 1 : 0;
  const int b      = blockIdx.x;

  // ---------------- stage inputs into LDS ----------------
  {
    const size_t kbase = (size_t)b * S_ * D_;
    for (int i = tid; i < S_ * D_; i += 128)
      keysLds[(i >> 6) * KPAD + (i & 63)] = keys[kbase + i];
    for (int i = tid; i < D_; i += 128) qLds[i] = query[(size_t)b * D_ + i];
    for (int i = tid; i < K1_ * H1_; i += 128) {
      int k = i >> 6, n = i & 63;
      w1Lds[n * APAD + k] = f2bf(W1[i]);
    }
    for (int i = tid; i < H1_ * H2_; i += 128) {
      int k = i >> 5, n = i & 31;
      w2Lds[n * W2PAD + k] = f2bf(W2[i]);
    }
    if (tid < 64) b1Lds[tid] = b1[tid];
    if (tid < 32) { b2Lds[tid] = b2[tid]; w3Lds[tid] = W3[tid]; }
  }
  const float A1 = a1[0], A2 = a2[0], B3 = b3[0];
  __syncthreads();

  // ---------------- hoist loop-invariant B-fragments into registers ----------
  const int nbase = wave * 16 + ln16;          // GEMM1 output column for this lane
  const int aoff = laneHi ? 8 : 0;             // A-matrix K sub-offset per half-wave
  const int boff = laneHi ? 16 : 0;            // unused; B fragments also load with aoff

  AFrag w1f[8];                                // W1 fragments, one per k-step
  #pragma unroll
  for (int kk = 0; kk < 8; ++kk) {
    const uint16_t* bp = w1Lds + nbase * APAD + kk * 32 + aoff;
    #pragma unroll
    for (int v = 0; v < 4; ++v) w1f[kk].u[v] = ld2bf(bp + 2 * v);
    #pragma unroll
    for (int v = 0; v < 4; ++v) w1f[kk].u[4 + v] = ld2bf(bp + 16 + 2 * v);
  }
  AFrag w2f[2];                                // W2 fragments (waves 0,1 only)
  if (wave < 2) {
    #pragma unroll
    for (int kk = 0; kk < 2; ++kk) {
      const uint16_t* bp = w2Lds + nbase * W2PAD + kk * 32 + aoff;
      #pragma unroll
      for (int v = 0; v < 4; ++v) w2f[kk].u[v] = ld2bf(bp + 2 * v);
      #pragma unroll
      for (int v = 0; v < 4; ++v) w2f[kk].u[4 + v] = ld2bf(bp + 16 + 2 * v);
    }
  }

  // A-tile build geometry (8 threads per row, 32 consecutive cols each; a
  // 32-aligned block never crosses a 64-aligned concat-region boundary)
  const int ar   = tid >> 3;                   // 0..15
  const int ac0  = (tid & 7) * 32;
  const int areg = ac0 >> 6;                   // which of [k | q | k*q | k-q]
  const int ajb  = ac0 & 63;

  // ---------------- per s-tile MLP via WMMA ----------------
  for (int t = 0; t < NT_; ++t) {
    const int s0 = t * 16;

    // Build att_in tile [16 x 256] bf16, two elems per 32-bit LDS store
    {
      const int srow = s0 + ar;
      const bool inb = srow < S_;
      const float* krow = &keysLds[srow * KPAD];
      uint32_t* dst = (uint32_t*)(attLds + ar * APAD + ac0);
      #pragma unroll
      for (int cc = 0; cc < 32; cc += 2) {
        int j = ajb + cc;
        float kv0 = inb ? krow[j] : 0.f;
        float kv1 = inb ? krow[j + 1] : 0.f;
        float q0 = qLds[j], q1 = qLds[j + 1];
        float v0, v1;
        if      (areg == 0) { v0 = kv0;       v1 = kv1;       }
        else if (areg == 1) { v0 = q0;        v1 = q1;        }
        else if (areg == 2) { v0 = kv0 * q0;  v1 = kv1 * q1;  }
        else                { v0 = kv0 - q0;  v1 = kv1 - q1;  }
        dst[cc >> 1] = (uint32_t)f2bf(v0) | ((uint32_t)f2bf(v1) << 16);
      }
    }
    __syncthreads();

    // GEMM1: [16 x 256] x [256 x 64]; only A-fragments come from LDS now
    v8f acc = {};
    const uint16_t* arow = attLds + ln16 * APAD;
    #pragma unroll
    for (int kk = 0; kk < 8; ++kk) {
      const int kb = kk * 32;
      AFrag a;
      #pragma unroll
      for (int v = 0; v < 4; ++v)
        a.u[v] = ld2bf(arow + kb + aoff + 2 * v);
      #pragma unroll
      for (int v = 0; v < 4; ++v)
        a.u[4 + v] = ld2bf(arow + kb + 16 + aoff + 2 * v);
      acc = __builtin_amdgcn_wmma_f32_16x16x32_bf16(false, a.v, false, w1f[kk].v,
                                                    (short)0, acc, false, false);
    }

    // bias + PReLU, restage h as bf16 [16 x 64]
    #pragma unroll
    for (int v = 0; v < 8; ++v) {
      int m = v + (laneHi ? 8 : 0);
      float x = acc[v] + b1Lds[nbase];
      float h = (x >= 0.f) ? x : A1 * x;
      hLds[m * HPAD + nbase] = f2bf(h);
    }
    __syncthreads();

    // GEMM2: [16 x 64] x [64 x 32] -> waves 0,1 own the two 16-col n-tiles
    if (wave < 2) {
      v8f acc2 = {};
      const uint16_t* hrow = hLds + ln16 * HPAD;
      #pragma unroll
      for (int kk = 0; kk < 2; ++kk) {
        const int kb = kk * 32;
        AFrag a;
        #pragma unroll
        for (int v = 0; v < 4; ++v)
          a.u[v] = ld2bf(hrow + kb + aoff + 2 * v);
        #pragma unroll
        for (int v = 0; v < 4; ++v)
          a.u[4 + v] = ld2bf(hrow + kb + 16 + aoff + 2 * v);
        acc2 = __builtin_amdgcn_wmma_f32_16x16x32_bf16(false, a.v, false, w2f[kk].v,
                                                       (short)0, acc2, false, false);
      }
      #pragma unroll
      for (int v = 0; v < 8; ++v) {
        int m = v + (laneHi ? 8 : 0);
        float x = acc2[v] + b2Lds[nbase];
        float h2 = (x >= 0.f) ? x : A2 * x;
        scTile[m * SCPAD + nbase] = h2;
      }
    }
    __syncthreads();

    // scores: 32-length dot with W3 + bias + mask
    if (tid < 16) {
      const int srow = s0 + tid;
      float s = B3;
      #pragma unroll
      for (int c = 0; c < H2_; ++c) s += scTile[tid * SCPAD + c] * w3Lds[c];
      bool ok = (srow < S_) && (kmask[(size_t)b * S_ + srow] != 0);
      scores[srow] = ok ? s : -__builtin_inff();
    }
    __syncthreads();
  }

  // ---------------- masked softmax over S ----------------
  float lmax = -__builtin_inff();
  for (int s = tid; s < S_; s += 128) lmax = fmaxf(lmax, scores[s]);
  red[tid] = lmax;
  __syncthreads();
  for (int off = 64; off > 0; off >>= 1) {
    if (tid < off) red[tid] = fmaxf(red[tid], red[tid + off]);
    __syncthreads();
  }
  const float mx = red[0];
  __syncthreads();

  float lsum = 0.f;
  for (int s = tid; s < S_; s += 128) {
    float sc = scores[s];
    float e = (sc == -__builtin_inff()) ? 0.f : __expf(sc - mx);
    scores[s] = e;
    lsum += e;
  }
  red[tid] = lsum;
  __syncthreads();
  for (int off = 64; off > 0; off >>= 1) {
    if (tid < off) red[tid] += red[tid + off];
    __syncthreads();
  }
  const float tot = red[0];
  __syncthreads();
  const float inv = (tot > 0.f) ? (1.f / tot) : 0.f;   // all-masked -> NaN -> 0 per reference

  for (int s = tid; s < S_; s += 128) {
    float w = scores[s] * inv;
    scores[s] = w;
    outW[(size_t)b * S_ + s] = w;
  }
  __syncthreads();

  // ---------------- weighted sum of keys ----------------
  {
    const int d = tid & 63;
    const int half = tid >> 6;          // 0 or 1: split S range
    float acc = 0.f;
    const int sBeg = half * 100, sEnd = sBeg + 100;
    for (int s = sBeg; s < sEnd; ++s) acc += scores[s] * keysLds[s * KPAD + d];
    red[tid] = acc;
    __syncthreads();
    if (tid < 64) outSum[(size_t)b * D_ + tid] = red[tid] + red[tid + 64];
  }
}

// ---------------- host-side launch ----------------
extern "C" void kernel_launch(void* const* d_in, const int* in_sizes, int n_in,
                              void* d_out, int out_size, void* d_ws, size_t ws_size,
                              hipStream_t stream) {
  (void)in_sizes; (void)n_in; (void)out_size; (void)d_ws; (void)ws_size;
  const float* query = (const float*)d_in[0];
  const float* keys  = (const float*)d_in[1];
  const int*   kmask = (const int*)  d_in[2];
  const float* W1 = (const float*)d_in[3];
  const float* b1 = (const float*)d_in[4];
  const float* a1 = (const float*)d_in[5];
  const float* W2 = (const float*)d_in[6];
  const float* b2 = (const float*)d_in[7];
  const float* a2 = (const float*)d_in[8];
  const float* W3 = (const float*)d_in[9];
  const float* b3 = (const float*)d_in[10];

  float* outSum = (float*)d_out;              // [B, D]
  float* outW   = outSum + (size_t)B_ * D_;   // [B, S]

  constexpr size_t smemBytes =
      (size_t)(S_ * KPAD + 16 * SCPAD + 208 + 64 + 64 + 32 + 32 + 128) * 4 +
      (size_t)(H1_ * APAD + H2_ * W2PAD + 16 * APAD + 16 * HPAD) * 2;   // = 107776

  din_attn_kernel<<<dim3(B_), dim3(128), smemBytes, stream>>>(
      query, keys, kmask, W1, b1, a1, W2, b2, a2, W3, b3, outSum, outW);
}
